// DSAttention_24558622998761
// MI455X (gfx1250) — matrix operand order repaired and checked by reading, not compiled, not run on hardware
//
#include <hip/hip_runtime.h>

// ---------------------------------------------------------------------------
// Flash-attention style causal attention for MI455X (gfx1250, wave32, WMMA).
// scores[b,h,l,s] = (Q.K * tau[b] + delta[b,s]) * (1/sqrt(E)), causal mask,
// softmax over s, then O = P.V.  B=4, L=2048, H=8, E=64.
//
// Phase 1 converts K -> f16 head-major Kh[b][h][s][e] and V -> f16 transposed
// Vt[b][h][e][s] so phase 2's WMMA B-operands are contiguous b128 loads.
// Softmax row-max uses fused v_max_num_f32_dpp butterflies (single VALU op
// per step, no LDS); row-sum rides the matrix pipe (P x ones via WMMA).
// ---------------------------------------------------------------------------

typedef __attribute__((ext_vector_type(16))) _Float16 v16h;
typedef __attribute__((ext_vector_type(8)))  _Float16 v8h;
typedef __attribute__((ext_vector_type(8)))  float    v8f;
typedef __attribute__((ext_vector_type(4)))  float    v4f;

#define WMMA_F16F32(a, b, c) \
  __builtin_amdgcn_wmma_f32_16x16x32_f16(false, (a), false, (b), (short)0, (c), false, false)

static constexpr int Bn = 4, L = 2048, H = 8, E = 64;
static constexpr float SCALE = 0.125f;                  // 1/sqrt(64)
static constexpr float LOG2E = 1.4426950408889634f;

// max across the 16-lane half-row (matches WMMA C-layout row grouping).
// Each step is ONE fused VALU op: dst = max(dpp16(src0), src1).
static __device__ __forceinline__ float redmax16_dpp(float v) {
  asm("v_max_num_f32_dpp %0, %0, %0 quad_perm:[1,0,3,2] row_mask:0xf bank_mask:0xf"
      : "+v"(v));                                        // lane ^ 1
  asm("v_max_num_f32_dpp %0, %0, %0 quad_perm:[2,3,0,1] row_mask:0xf bank_mask:0xf"
      : "+v"(v));                                        // lane ^ 2
  asm("v_max_num_f32_dpp %0, %0, %0 row_half_mirror row_mask:0xf bank_mask:0xf"
      : "+v"(v));                                        // lane ^ 7 (quads uniform)
  asm("v_max_num_f32_dpp %0, %0, %0 row_mirror row_mask:0xf bank_mask:0xf"
      : "+v"(v));                                        // lane ^ 15 (octets uniform)
  return v;
}

static __device__ __forceinline__ v8h load8h(const float* __restrict__ p) {
  v4f a = *(const v4f*)p;
  v4f b = *(const v4f*)(p + 4);
  v8h r;
  r[0] = (_Float16)a[0]; r[1] = (_Float16)a[1]; r[2] = (_Float16)a[2]; r[3] = (_Float16)a[3];
  r[4] = (_Float16)b[0]; r[5] = (_Float16)b[1]; r[6] = (_Float16)b[2]; r[7] = (_Float16)b[3];
  return r;
}
static __device__ __forceinline__ v16h cat8(v8h lo, v8h hi) {
  return __builtin_shufflevector(lo, hi, 0, 1, 2, 3, 4, 5, 6, 7, 8, 9, 10, 11, 12, 13, 14, 15);
}

// ---------------------------------------------------------------------------
// Phase 1: K -> Kh[b][h][s][e] (f16), V -> Vt[b][h][e][s] (f16, transposed)
// ---------------------------------------------------------------------------
__global__ __launch_bounds__(256) void dsattn_prep_kernel(
    const float* __restrict__ K, const float* __restrict__ V,
    _Float16* __restrict__ Kh, _Float16* __restrict__ Vt) {
  constexpr int TS = 64;
  const int t = threadIdx.x;
  const int nt = L / TS;
  const int tile = blockIdx.x % nt;
  const int h = (blockIdx.x / nt) % H;
  const int b = blockIdx.x / (nt * H);
  const int s0 = tile * TS;

  __shared__ _Float16 lds[TS][E + 2];                   // bank-skew pad
#pragma unroll
  for (int k = 0; k < 16; ++k) {
    const int idx = t + 256 * k;
    const int s = idx >> 6, e = idx & 63;               // coalesced along e
    const size_t gi = (((size_t)b * L + s0 + s) * H + h) * E + e;
    Kh[(((size_t)b * H + h) * L + s0 + s) * E + e] = (_Float16)K[gi];
    lds[s][e] = (_Float16)V[gi];
  }
  __syncthreads();
#pragma unroll
  for (int k = 0; k < 16; ++k) {
    const int idx = t + 256 * k;
    const int e = idx >> 6, s = idx & 63;               // coalesced along s
    Vt[(((size_t)b * H + h) * E + e) * L + s0 + s] = lds[s][e];
  }
}

// ---------------------------------------------------------------------------
// Phase 2: one wave = one 16-row query tile; 32 keys per iteration.
// ---------------------------------------------------------------------------
template <bool MASKED>
static __device__ __forceinline__ void attn_tile(
    int s_base, int q_base, int m_lane, int hi, int r0,
    const _Float16* __restrict__ khbh, const _Float16* __restrict__ vtbh,
    const float* __restrict__ delta_b, float st2,
    const v16h& aQ0, const v16h& aQ1, const v16h& bOnes,
    v8f& O0, v8f& O1, v8f& O2, v8f& O3, v8f& Lacc,
    float (&Mrow)[8], _Float16 (*pw)[32]) {
  // gfx1250 prefetch of next tile's K row / V run (speculative, may be OOB)
  __builtin_prefetch(khbh + (size_t)(s_base + 32 + m_lane) * E, 0, 3);
  __builtin_prefetch(vtbh + (size_t)m_lane * L + s_base + 32, 0, 3);

  // ---- S = Q . K^T (16 x 32 keys): B-operand = 16 contiguous e per lane ---
  v8f S[2];
#pragma unroll
  for (int g = 0; g < 2; ++g) {
    const _Float16* krow = khbh + (size_t)(s_base + 16 * g + m_lane) * E;
    const v16h bK0 = *(const v16h*)(krow + 16 * hi);
    const v16h bK1 = *(const v16h*)(krow + 32 + 16 * hi);
    v8f c = WMMA_F16F32(aQ0, bK0, (v8f){});
    S[g] = WMMA_F16F32(aQ1, bK1, c);
  }

  // ---- online softmax (log2 domain); row-max via fused DPP max ------------
  const float sd0 = (SCALE * LOG2E) * delta_b[s_base + m_lane];
  const float sd1 = (SCALE * LOG2E) * delta_b[s_base + 16 + m_lane];
  const int s0 = s_base + m_lane;
  const int s1 = s0 + 16;

  float p0[8], p1[8], alpha[8];
#pragma unroll
  for (int r = 0; r < 8; ++r) {
    float a0 = st2 * S[0][r] + sd0;
    float a1 = st2 * S[1][r] + sd1;
    if (MASKED) {
      const int lg = q_base + r + 8 * hi;
      a0 = (s0 <= lg) ? a0 : -1e30f;
      a1 = (s1 <= lg) ? a1 : -1e30f;
    }
    const float rmax = redmax16_dpp(fmaxf(a0, a1));
    const float mnew = fmaxf(Mrow[r], rmax);
    alpha[r] = exp2f(Mrow[r] - mnew);
    Mrow[r] = mnew;
    p0[r] = exp2f(a0 - mnew);
    p1[r] = exp2f(a1 - mnew);
    O0[r] *= alpha[r]; O1[r] *= alpha[r]; O2[r] *= alpha[r]; O3[r] *= alpha[r];
    Lacc[r] *= alpha[r];
  }

  // ---- stage P (f16) in per-wave LDS, row-major [m][s_local] --------------
#pragma unroll
  for (int r = 0; r < 8; ++r) {
    pw[r + 8 * hi][m_lane]      = (_Float16)p0[r];
    pw[r + 8 * hi][16 + m_lane] = (_Float16)p1[r];
  }
  asm volatile("s_wait_dscnt 0" ::: "memory");           // same-wave DS RAW fence

  // ---- re-read P as A operand (contiguous K runs in row-major tile) -------
  const v16h aP = cat8(*(const v8h*)&pw[m_lane][2 * r0],
                       *(const v8h*)&pw[m_lane][8 + 2 * r0]);

  // ---- row-sum via matrix pipe: Lacc += P . ones(32x16) -------------------
  Lacc = WMMA_F16F32(aP, bOnes, Lacc);

  // ---- O += P . V : B-operand = 16 contiguous s per lane from Vt ----------
#pragma unroll
  for (int j = 0; j < 4; ++j) {
    const v16h bV = *(const v16h*)(vtbh + (size_t)(16 * j + m_lane) * L +
                                   s_base + 16 * hi);
    if (j == 0) O0 = WMMA_F16F32(aP, bV, O0);
    if (j == 1) O1 = WMMA_F16F32(aP, bV, O1);
    if (j == 2) O2 = WMMA_F16F32(aP, bV, O2);
    if (j == 3) O3 = WMMA_F16F32(aP, bV, O3);
  }
}

__global__ __launch_bounds__(256) void dsattn_fwd_kernel(
    const float* __restrict__ Q, const _Float16* __restrict__ Kh,
    const _Float16* __restrict__ Vt, const float* __restrict__ tau,
    const float* __restrict__ delta, float* __restrict__ out) {
  constexpr int QT_PER_BLOCK = 8;                        // 8 waves * 16 rows

  const int lane   = threadIdx.x & 31;
  const int wave   = threadIdx.x >> 5;
  const int m_lane = lane & 15;
  const int hi     = lane >> 4;
  const int r0     = hi * 8;

  const int nQT = L / (16 * QT_PER_BLOCK);
  const int qtb = blockIdx.x % nQT;
  const int h   = (blockIdx.x / nQT) % H;
  const int b   = blockIdx.x / (nQT * H);
  const int q_base = qtb * (16 * QT_PER_BLOCK) + wave * 16;

  const float st2 = SCALE * LOG2E * tau[b];
  const _Float16* khbh = Kh + ((size_t)b * H + h) * L * E;
  const _Float16* vtbh = Vt + ((size_t)b * H + h) * E * L;
  const float* delta_b = delta + (size_t)b * L;

  // Q tile (16x64) as two A operands (contiguous K runs {2*r0..2*r0+15})
  const float* qrow = Q + (((size_t)b * L + (q_base + m_lane)) * H + h) * E;
  const v16h aQ0 = cat8(load8h(qrow + 0  + 2 * r0), load8h(qrow + 8 + 2 * r0));
  const v16h aQ1 = cat8(load8h(qrow + 32 + 2 * r0), load8h(qrow + 40 + 2 * r0));

  v16h bOnes;
#pragma unroll
  for (int i = 0; i < 16; ++i) bOnes[i] = (_Float16)1.0f;

  v8f O0 = {}, O1 = {}, O2 = {}, O3 = {}, Lacc = {};
  float Mrow[8];
#pragma unroll
  for (int r = 0; r < 8; ++r) Mrow[r] = -1e30f;

  __shared__ __align__(16) _Float16 plds[QT_PER_BLOCK][16][32];
  _Float16(*pw)[32] = plds[wave];

  int s_base = 0;
  for (; s_base + 31 <= q_base; s_base += 32)            // fully unmasked tiles
    attn_tile<false>(s_base, q_base, m_lane, hi, r0, khbh, vtbh, delta_b, st2,
                     aQ0, aQ1, bOnes, O0, O1, O2, O3, Lacc, Mrow, pw);
  for (; s_base <= q_base + 15; s_base += 32)            // diagonal (masked)
    attn_tile<true>(s_base, q_base, m_lane, hi, r0, khbh, vtbh, delta_b, st2,
                    aQ0, aQ1, bOnes, O0, O1, O2, O3, Lacc, Mrow, pw);

  // ---- normalize and store (B,L,H,E) --------------------------------------
#pragma unroll
  for (int r = 0; r < 8; ++r) {
    const float inv = __builtin_amdgcn_rcpf(Lacc[r]);
    const int lg = q_base + r + 8 * hi;
    float* orow = out + (((size_t)b * L + lg) * H + h) * E + m_lane;
    orow[0]  = O0[r] * inv;
    orow[16] = O1[r] * inv;
    orow[32] = O2[r] * inv;
    orow[48] = O3[r] * inv;
  }
}

extern "C" void kernel_launch(void* const* d_in, const int* in_sizes, int n_in,
                              void* d_out, int out_size, void* d_ws, size_t ws_size,
                              hipStream_t stream) {
  (void)in_sizes; (void)n_in; (void)out_size; (void)ws_size;
  const float* Q     = (const float*)d_in[0];
  const float* K     = (const float*)d_in[1];
  const float* V     = (const float*)d_in[2];
  // d_in[3] = attn_mask (bool, causal) -- computed analytically
  const float* tau   = (const float*)d_in[4];
  const float* delta = (const float*)d_in[5];
  float* out = (float*)d_out;

  // workspace: Kh (f16, B*H*L*E) then Vt (f16, B*H*E*L)  -> 16.8 MB total
  _Float16* Kh = (_Float16*)d_ws;
  _Float16* Vt = Kh + (size_t)Bn * H * L * E;

  dsattn_prep_kernel<<<Bn * H * (L / 64), 256, 0, stream>>>(K, V, Kh, Vt);
  dsattn_fwd_kernel<<<Bn * H * (L / 128), 256, 0, stream>>>(Q, Kh, Vt, tau, delta, out);
}
